// MHA_68831145886222
// MI455X (gfx1250) — matrix operand order repaired and checked by reading, not compiled, not run on hardware
//
#include <hip/hip_runtime.h>

typedef unsigned short u16;
typedef unsigned int   u32;

typedef __bf16 v16bf __attribute__((ext_vector_type(16)));
typedef float  v8f   __attribute__((ext_vector_type(8)));

union BF16Frag { v16bf v; uint4 q[2]; };

__device__ __forceinline__ u16 f2bf(float f) {
    u32 u = __float_as_uint(f);
    u32 r = (u + 0x7fffu + ((u >> 16) & 1u)) >> 16;   // round-to-nearest-even
    return (u16)r;
}

__device__ __forceinline__ v8f wmma_bf16(const BF16Frag& a, const BF16Frag& b, v8f c) {
    // D = A(16x32 bf16) * B(32x16 bf16) + C(16x16 f32)
    return __builtin_amdgcn_wmma_f32_16x16x32_bf16(false, a.v, false, b.v,
                                                   (short)0, c, false, false);
}

// ---- CDNA5 async global->LDS copy (ASYNCcnt) with portable fallback -------
// Builtin signature (from hipcc diagnostic): first arg is
// 'int __vector(4) __device__ *' i.e. addrspace(1) int4*; LDS side addrspace(3).
typedef int v4i __attribute__((vector_size(16)));
typedef __attribute__((address_space(1))) v4i GV4;
typedef __attribute__((address_space(3))) v4i LV4;

__device__ __forceinline__ void async_cp16(const void* g, void* l) {
#if __has_builtin(__builtin_amdgcn_global_load_async_to_lds_b128)
    __builtin_amdgcn_global_load_async_to_lds_b128((GV4*)g, (LV4*)l, 0, 0);
#else
    *(uint4*)l = *(const uint4*)g;
#endif
}

__device__ __forceinline__ void wait_async0() {
#if __has_builtin(__builtin_amdgcn_s_wait_asynccnt)
    __builtin_amdgcn_s_wait_asynccnt(0);
#else
    asm volatile("s_wait_asynccnt 0x0" ::: "memory");
#endif
}

__device__ __forceinline__ void wait_ds0() {
#if __has_builtin(__builtin_amdgcn_s_wait_dscnt)
    __builtin_amdgcn_s_wait_dscnt(0);
#else
    asm volatile("s_wait_dscnt 0x0" ::: "memory");
#endif
}

// ---------------------------------------------------------------------------
// Bulk f32 -> bf16 conversion (one float4 per thread).
// ---------------------------------------------------------------------------
__global__ __launch_bounds__(256) void cvt_bf16(const float* __restrict__ in,
                                                u16* __restrict__ out, int n4)
{
    int i = blockIdx.x * 256 + threadIdx.x;
    if (i < n4) {
        float4 f = ((const float4*)in)[i];
        uint2 pk;
        pk.x = (u32)f2bf(f.x) | ((u32)f2bf(f.y) << 16);
        pk.y = (u32)f2bf(f.z) | ((u32)f2bf(f.w) << 16);
        ((uint2*)out)[i] = pk;
    }
}

// ---------------------------------------------------------------------------
// GEMM: C[M,N] (f32) = A[M,K] bf16 row-major * B[N,K]^T bf16 (weights).
// Block tile 128x64, 8 waves each owning 32x32 (2x2 WMMA accumulators).
// Double-buffered async global->LDS staging; K stepped by 32.
// ---------------------------------------------------------------------------
__global__ __launch_bounds__(256) void gemm_bf16(const u16* __restrict__ A,
                                                 const u16* __restrict__ B,
                                                 float* __restrict__ C,
                                                 int N, int K)
{
    __shared__ __align__(16) u16 As[2][128][32];
    __shared__ __align__(16) u16 Bs[2][64][32];

    const int tid   = threadIdx.x;
    const int lane  = tid & 31;
    const int wv    = tid >> 5;          // 0..7
    const int wrow  = (wv & 3) * 32;
    const int wcol  = (wv >> 2) * 32;
    const int lrow  = lane & 15;
    const int lhalf = lane >> 4;

    const int m0 = blockIdx.y * 128;
    const int n0 = blockIdx.x * 64;

    auto stage = [&](int k0, int b) {
        #pragma unroll
        for (int i = 0; i < 2; ++i) {                       // A tile: 512 x 16B
            int seg = tid * 2 + i;
            int row = seg >> 2, ks = (seg & 3) * 8;
            async_cp16(A + (size_t)(m0 + row) * K + k0 + ks, &As[b][row][ks]);
        }
        {                                                   // B tile: 256 x 16B
            int row = tid >> 2, ks = (tid & 3) * 8;
            async_cp16(B + (size_t)(n0 + row) * K + k0 + ks, &Bs[b][row][ks]);
        }
    };

    v8f acc[2][2] = {};

    stage(0, 0);
    wait_async0();
    __syncthreads();

    int buf = 0;
    for (int k0 = 0; k0 < K; k0 += 32) {
        if (k0 + 32 < K) stage(k0 + 32, buf ^ 1);   // prefetch next K-slab

        BF16Frag a[2], b[2];
        #pragma unroll
        for (int mr = 0; mr < 2; ++mr) {
            const u16* p = &As[buf][wrow + mr*16 + lrow][lhalf*8];
            a[mr].q[0] = *(const uint4*)p;          // K = h*8 .. +8
            a[mr].q[1] = *(const uint4*)(p + 16);   // K = 16+h*8 .. +8
        }
        #pragma unroll
        for (int nc = 0; nc < 2; ++nc) {
            const u16* p = &Bs[buf][wcol + nc*16 + lrow][lhalf*8];
            b[nc].q[0] = *(const uint4*)p;          // K = h*8 .. +8
            b[nc].q[1] = *(const uint4*)(p + 16);
        }
        #pragma unroll
        for (int mr = 0; mr < 2; ++mr)
            #pragma unroll
            for (int nc = 0; nc < 2; ++nc)
                acc[mr][nc] = wmma_bf16(a[mr], b[nc], acc[mr][nc]);

        wait_async0();
        __syncthreads();
        buf ^= 1;
    }

    #pragma unroll
    for (int mr = 0; mr < 2; ++mr)
        #pragma unroll
        for (int nc = 0; nc < 2; ++nc)
            #pragma unroll
            for (int r = 0; r < 8; ++r) {
                int row = m0 + wrow + mr*16 + r + 8*lhalf;
                int col = n0 + wcol + nc*16 + lrow;
                C[(size_t)row * N + col] = acc[mr][nc][r];
            }
}

// ---------------------------------------------------------------------------
// RMSNorm + RoPE + bf16 pack. One wave per 128-wide head vector.
// Q pre-scaled by 1/sqrt(HD); V written TRANSPOSED: vtb[(g*128+d)*2048 + s].
// ---------------------------------------------------------------------------
__global__ __launch_bounds__(128) void norm_rope(const float* __restrict__ qkv,
                                                 const float* __restrict__ qw,
                                                 const float* __restrict__ kw,
                                                 u16* __restrict__ qb,
                                                 u16* __restrict__ kb,
                                                 u16* __restrict__ vtb)
{
    const int lane = threadIdx.x & 31;
    const int vec  = blockIdx.x * 4 + (threadIdx.x >> 5);
    const int s = vec / 48;
    const int r = vec % 48;

    const float* src; const float* w = nullptr;
    float qscale = 1.0f;
    if (r < 32)      { src = qkv + (size_t)s*6144 + r*128;           w = qw;
                       qscale = 0.08838834764831845f; }              // 1/sqrt(128)
    else if (r < 40) { src = qkv + (size_t)s*6144 + 4096 + (r-32)*128; w = kw; }
    else             { src = qkv + (size_t)s*6144 + 5120 + (r-40)*128; }

    float4 x = *(const float4*)(src + lane*4);
    float y0 = x.x, y1 = x.y, y2 = x.z, y3 = x.w;

    if (r < 40) {   // RMSNorm + RoPE for q and k
        float ss = y0*y0 + y1*y1 + y2*y2 + y3*y3;
        #pragma unroll
        for (int off = 1; off < 32; off <<= 1) ss += __shfl_xor(ss, off, 32);
        float inv = rsqrtf(ss * (1.0f/128.0f) + 1e-5f);
        float4 wl = *(const float4*)(w + lane*4);
        y0 *= inv * wl.x; y1 *= inv * wl.y; y2 *= inv * wl.z; y3 *= inv * wl.w;

        const int j0 = lane * 2;                 // RoPE pair index (0..63)
        float f0 = __powf(1.0e6f, -(float)(j0    ) * (1.0f/64.0f));
        float f1 = __powf(1.0e6f, -(float)(j0 + 1) * (1.0f/64.0f));
        float c0, s0, c1, s1;
        __sincosf((float)s * f0, &s0, &c0);
        __sincosf((float)s * f1, &s1, &c1);
        float a0 = y0*c0 - y1*s0, b0 = y0*s0 + y1*c0;
        float a1 = y2*c1 - y3*s1, b1 = y2*s1 + y3*c1;
        y0 = a0 * qscale; y1 = b0 * qscale; y2 = a1 * qscale; y3 = b1 * qscale;

        u16* dst = (r < 32) ? (qb + (size_t)s*4096 + r*128)
                            : (kb + (size_t)s*1024 + (r-32)*128);
        uint2 pk;
        pk.x = (u32)f2bf(y0) | ((u32)f2bf(y1) << 16);
        pk.y = (u32)f2bf(y2) | ((u32)f2bf(y3) << 16);
        *(uint2*)(dst + lane*4) = pk;
    } else {        // V: transpose into [g*128+d][s]
        const int g = r - 40;
        const int d0 = g*128 + lane*4;
        vtb[(size_t)(d0 + 0)*2048 + s] = f2bf(y0);
        vtb[(size_t)(d0 + 1)*2048 + s] = f2bf(y1);
        vtb[(size_t)(d0 + 2)*2048 + s] = f2bf(y2);
        vtb[(size_t)(d0 + 3)*2048 + s] = f2bf(y3);
    }
}

// ---------------------------------------------------------------------------
// Flash attention (causal, GQA). Block = 4 waves = 64 query rows of one head.
// Double-buffered async global->LDS staging of K (row-major) and V^T.
// QK^T + PV via bf16 WMMA; online softmax; P goes C-layout -> LDS -> A-layout.
// ---------------------------------------------------------------------------
__global__ __launch_bounds__(128) void attn_fwd(const u16* __restrict__ qb,
                                                const u16* __restrict__ kb,
                                                const u16* __restrict__ vtb,
                                                u16* __restrict__ ob)
{
    __shared__ __align__(16) u16 Ks[2][32][128];   // [t][d]
    __shared__ __align__(16) u16 Vt[2][128][32];   // [d][t]
    __shared__ __align__(16) u16 Ps[4][16][32];    // per-wave P staging

    const int tid = threadIdx.x, lane = tid & 31, wv = tid >> 5;
    const int lrow = lane & 15, lhalf = lane >> 4;
    const int h = blockIdx.y, g = h >> 2;           // HG = 4
    const int q0 = blockIdx.x * 64;

    auto stage = [&](int tb, int b) {
        const int t0 = tb * 32;
        #pragma unroll
        for (int i = 0; i < 4; ++i) {               // K tile: 512 x 16B
            int seg = tid * 4 + i;
            int tr = seg >> 4, ds_ = (seg & 15) * 8;
            async_cp16(kb + (size_t)(t0 + tr)*1024 + g*128 + ds_, &Ks[b][tr][ds_]);
        }
        #pragma unroll
        for (int i = 0; i < 4; ++i) {               // V^T tile: 512 x 16B
            int seg = tid * 4 + i;
            int d = seg >> 2, ts = (seg & 3) * 8;
            async_cp16(vtb + (size_t)(g*128 + d)*2048 + t0 + ts, &Vt[b][d][ts]);
        }
    };

    // persistent Q fragments: 16 rows x 128 K (4 chunks of 32)
    BF16Frag qf[4];
    {
        const u16* qp = qb + (size_t)(q0 + wv*16 + lrow) * 4096 + h*128;
        #pragma unroll
        for (int c = 0; c < 4; ++c) {
            qf[c].q[0] = *(const uint4*)(qp + c*32 + lhalf*8);
            qf[c].q[1] = *(const uint4*)(qp + c*32 + 16 + lhalf*8);
        }
    }

    v8f O[8] = {};
    float m[8], l[8];
    #pragma unroll
    for (int r = 0; r < 8; ++r) { m[r] = -1.0e30f; l[r] = 0.0f; }

    const int ntb = (q0 + 64) / 32;                 // key tiles under causal mask

    stage(0, 0);
    wait_async0();
    __syncthreads();

    int buf = 0;
    for (int tb = 0; tb < ntb; ++tb) {
        const int t0 = tb * 32;
        if (tb + 1 < ntb) stage(tb + 1, buf ^ 1);   // prefetch next key tile

        // --- scores: S(16x32) = Q(16x128) * K^T ---
        v8f sc[2];
        #pragma unroll
        for (int nt = 0; nt < 2; ++nt) {
            v8f sacc = {};
            #pragma unroll
            for (int c = 0; c < 4; ++c) {
                BF16Frag b;
                const u16* p = &Ks[buf][nt*16 + lrow][c*32 + lhalf*8];
                b.q[0] = *(const uint4*)p;
                b.q[1] = *(const uint4*)(p + 16);
                sacc = wmma_bf16(qf[c], b, sacc);
            }
            sc[nt] = sacc;
        }

        // --- causal mask + online softmax ---
        const int col0 = t0 + lrow, col1 = t0 + 16 + lrow;
        #pragma unroll
        for (int r = 0; r < 8; ++r) {
            const int row = q0 + wv*16 + r + 8*lhalf;
            float s0 = (col0 <= row) ? sc[0][r] : -1.0e30f;
            float s1 = (col1 <= row) ? sc[1][r] : -1.0e30f;
            float rm = fmaxf(s0, s1);
            #pragma unroll
            for (int off = 1; off < 16; off <<= 1) rm = fmaxf(rm, __shfl_xor(rm, off, 32));
            float mn = fmaxf(m[r], rm);
            float alpha = __expf(m[r] - mn);
            m[r] = mn;
            float p0 = __expf(s0 - mn);
            float p1 = __expf(s1 - mn);
            float rs = p0 + p1;
            #pragma unroll
            for (int off = 1; off < 16; off <<= 1) rs += __shfl_xor(rs, off, 32);
            l[r] = l[r] * alpha + rs;
            #pragma unroll
            for (int j = 0; j < 8; ++j) O[j][r] *= alpha;
            const int rl = r + 8*lhalf;
            Ps[wv][rl][lrow]      = f2bf(p0);
            Ps[wv][rl][16 + lrow] = f2bf(p1);
        }
        wait_ds0();   // same-wave LDS round trip for layout change

        // --- P fragment (A-layout) + PV ---
        BF16Frag pf;
        pf.q[0] = *(const uint4*)&Ps[wv][lrow][lhalf*8];
        pf.q[1] = *(const uint4*)&Ps[wv][lrow][16 + lhalf*8];
        #pragma unroll
        for (int j = 0; j < 8; ++j) {
            BF16Frag b;
            const u16* p = &Vt[buf][j*16 + lrow][lhalf*8];
            b.q[0] = *(const uint4*)p;
            b.q[1] = *(const uint4*)(p + 16);
            O[j] = wmma_bf16(pf, b, O[j]);
        }

        wait_async0();
        __syncthreads();
        buf ^= 1;
    }

    // --- normalize and store bf16 attention output (s, h, d) ---
    #pragma unroll
    for (int j = 0; j < 8; ++j) {
        #pragma unroll
        for (int r = 0; r < 8; ++r) {
            const int row = q0 + wv*16 + r + 8*lhalf;
            const int d   = j*16 + lrow;
            ob[(size_t)row * 4096 + h*128 + d] = f2bf(O[j][r] / l[r]);
        }
    }
}

// ---------------------------------------------------------------------------
extern "C" void kernel_launch(void* const* d_in, const int* in_sizes, int n_in,
                              void* d_out, int out_size, void* d_ws, size_t ws_size,
                              hipStream_t stream)
{
    (void)in_sizes; (void)n_in; (void)out_size; (void)ws_size;
    const float* x     = (const float*)d_in[0];
    const float* w_qkv = (const float*)d_in[1];
    const float* w_out = (const float*)d_in[2];
    const float* q_ln  = (const float*)d_in[3];
    const float* k_ln  = (const float*)d_in[4];
    float* out = (float*)d_out;

    // Workspace layout (184 MiB total):
    //   qkv   f32  [2048][6144] : 48 MiB   @ 0
    //   xb    bf16 [2048][4096] : 16 MiB   @ 48M
    //   wqkvb bf16 [6144][4096] : 48 MiB   @ 64M
    //   woutb bf16 [4096][4096] : 32 MiB   @ 112M
    //   qb    bf16 [2048][4096] : 16 MiB   @ 144M
    //   kbb   bf16 [2048][1024] :  4 MiB   @ 160M
    //   vtb   bf16 [1024][2048] :  4 MiB   @ 164M (transposed V)
    //   ab    bf16 [2048][4096] : 16 MiB   @ 168M
    char* ws = (char*)d_ws;
    float* qkv  = (float*)ws;
    u16* xb    = (u16*)(ws + (size_t)48*1024*1024);
    u16* wqkvb = (u16*)(ws + (size_t)64*1024*1024);
    u16* woutb = (u16*)(ws + (size_t)112*1024*1024);
    u16* qb    = (u16*)(ws + (size_t)144*1024*1024);
    u16* kbb   = (u16*)(ws + (size_t)160*1024*1024);
    u16* vtb   = (u16*)(ws + (size_t)164*1024*1024);
    u16* ab    = (u16*)(ws + (size_t)168*1024*1024);

    // 0) one-time f32 -> bf16 conversions (stream once at HBM rate)
    cvt_bf16<<<(2048*4096/4 + 255)/256, 256, 0, stream>>>(x,     xb,    2048*4096/4);
    cvt_bf16<<<(6144*4096/4 + 255)/256, 256, 0, stream>>>(w_qkv, wqkvb, 6144*4096/4);
    cvt_bf16<<<(4096*4096/4 + 255)/256, 256, 0, stream>>>(w_out, woutb, 4096*4096/4);

    // 1) QKV projection: qkv[s,e] = sum_d x[s,d] * w_qkv[e,d]  (f32 out)
    gemm_bf16<<<dim3(6144/64, 2048/128), 256, 0, stream>>>(xb, wqkvb, qkv, 6144, 4096);
    // 2) RMSNorm + RoPE + bf16 pack (q scaled by 1/sqrt(HD)); V transposed
    norm_rope<<<(2048*48)/4, 128, 0, stream>>>(qkv, q_ln, k_ln, qb, kbb, vtb);
    // 3) Causal GQA flash attention
    attn_fwd<<<dim3(2048/64, 32), 128, 0, stream>>>(qb, kbb, vtb, ab);
    // 4) Output projection: out[s,e] = sum_d attn[s,d] * w_out[e,d]
    gemm_bf16<<<dim3(4096/64, 2048/128), 256, 0, stream>>>(ab, woutb, out, 4096, 4096);
}
